// DeepSeekV3Mini_7980049236628
// MI455X (gfx1250) — matrix operand order repaired and checked by reading, not compiled, not run on hardware
//
#include <hip/hip_runtime.h>
#include <hip/hip_bf16.h>

// ---------------------------------------------------------------------------
// Model constants
// ---------------------------------------------------------------------------
constexpr int Bc  = 2;
constexpr int Sc  = 1024;
constexpr int Vc  = 32000;
constexpr int Dc  = 768;
constexpr int Hc  = 12;
constexpr int DFFc= 3072;
constexpr int Ec  = 8;
constexpr int DLc = 192;
constexpr int Lc  = 2;
constexpr int HDc = 64;

#define DEV __device__ __forceinline__

typedef __attribute__((ext_vector_type(16))) __bf16 v16bf;
typedef __attribute__((ext_vector_type(8)))  float  v8f;

union FragBF { v16bf v; unsigned short u[16]; };
union U4BF   { uint4 v; unsigned short u[8]; };

// f32 -> bf16 bits, round-to-nearest-even
DEV unsigned short f2bf(float f) {
    unsigned int u = __builtin_bit_cast(unsigned int, f);
    unsigned int r = u + 0x7FFFu + ((u >> 16) & 1u);
    return (unsigned short)(r >> 16);
}

DEV unsigned int f2bf2(float a, float b) {
    return (unsigned int)f2bf(a) | ((unsigned int)f2bf(b) << 16);
}

DEV v8f wmma_bf16(v16bf a, v16bf b, v8f c) {
    return __builtin_amdgcn_wmma_f32_16x16x32_bf16(false, a, false, b,
                                                   (short)0, c, false, false);
}

// Async global -> LDS copy, 16 bytes per lane (GV mode), tracked by ASYNCcnt.
// lds_addr: LDS byte address (low 32 bits of flat pointer = LDS offset).
DEV void async_g2l_b128(unsigned lds_addr, const void* gsrc) {
    asm volatile("global_load_async_to_lds_b128 %0, %1, off"
                 :: "v"(lds_addr), "v"(gsrc) : "memory");
}
DEV void wait_asynccnt0() {
    asm volatile("s_wait_asynccnt 0x0" ::: "memory");
}

// A fragment: 16x32 bf16 tile, row-major in LDS with leading dim ld (u16 units).
DEV v16bf load_A_frag(const unsigned short* p, int ld) {
    const int lane = threadIdx.x & 31;
    const int row  = lane & 15;
    const int kb   = (lane < 16) ? 0 : 16;
    FragBF f;
#pragma unroll
    for (int i = 0; i < 16; ++i) {
        f.u[i]     = p[row * ld + kb + i];
    }
    return f.v;
}

// B fragment from an N-major (transposed) tile: p[n*ld + k]. 16 contiguous u16.
DEV v16bf load_BT_frag(const unsigned short* p, int ld) {
    const int lane = threadIdx.x & 31;
    const int col  = lane & 15;
    const int kb   = (lane < 16) ? 0 : 16;
    FragBF f;
#pragma unroll
    for (int i = 0; i < 16; ++i) f.u[i] = p[col * ld + kb + i];
    return f.v;
}

// B fragment for K^T in attention (ksm row-major [key][64]).
DEV v16bf load_KT_frag(const unsigned short* ksm, int t, int koff) {
    const int lane = threadIdx.x & 31;
    const int key  = (t << 4) + (lane & 15);
    const int kb   = koff + ((lane < 16) ? 0 : 16);
    FragBF f;
#pragma unroll
    for (int i = 0; i < 16; ++i) f.u[i] = ksm[key * 64 + kb + i];
    return f.v;
}

// ---------------------------------------------------------------------------
// f32 -> bf16 streaming conversion (8 elements / thread, fully vectorized)
// ---------------------------------------------------------------------------
__global__ void cvt_kernel(const float* __restrict__ in,
                           unsigned short* __restrict__ out, int n8)
{
    int idx = blockIdx.x * blockDim.x + threadIdx.x;
    if (idx >= n8) return;
    const float4* p = (const float4*)(in + (size_t)idx * 8);
    float4 a = p[0], b = p[1];
    uint4 o;
    o.x = f2bf2(a.x, a.y); o.y = f2bf2(a.z, a.w);
    o.z = f2bf2(b.x, b.y); o.w = f2bf2(b.z, b.w);
    *(uint4*)(out + (size_t)idx * 8) = o;
}

// ---------------------------------------------------------------------------
// Generic WMMA GEMM on bf16 inputs:  C(MxN) = epilogue(A(MxK) @ B(KxN))
// 128 threads (4 waves). Block tile 128(M) x 64(N); each wave computes a
// 32x64 slab (8 v8f accumulators). K-stage 64 -> 16 wmma per wave/stage.
// A tile staged via async global->LDS DMA (ASYNCcnt); B tile via transpose
// scatter (needs 16-bit reshuffle the async engine cannot do).
// EPI: 0 = plain, 1 = gelu(acc+bias), 2 = resid+acc, 3 = C += gate*(acc+bias)
// ---------------------------------------------------------------------------
constexpr int BTLD = 72;   // padded N-major B tile stride (u16); rows 16B aligned

template <int EPI>
__global__ __launch_bounds__(128) void gemm_kernel(
    const unsigned short* __restrict__ A, const unsigned short* __restrict__ Bw,
    float* __restrict__ C, int M, int N, int Kd,
    const float* __restrict__ bias, const float* __restrict__ resid,
    const float* __restrict__ gate, int gateIdx)
{
    __shared__ unsigned short As[128 * 64];      // row-major [m][k]
    __shared__ unsigned short BsT[64 * BTLD];    // N-major  [n][k]

    const int tid  = threadIdx.x;
    const int lane = tid & 31;
    const int w    = tid >> 5;
    const int nb   = blockIdx.x * 64;
    const int mb   = blockIdx.y * 128;

    v8f acc0[4] = {};   // rows 32w   .. 32w+15
    v8f acc1[4] = {};   // rows 32w+16.. 32w+31

    for (int k0 = 0; k0 < Kd; k0 += 64) {
        __syncthreads();
        // A tile 128x64 bf16: async DMA, 16B per lane, 8 issues per thread
#pragma unroll
        for (int i = 0; i < 8; ++i) {
            int g  = tid + i * 128;
            int r  = g >> 3;
            int c8 = (g & 7) * 8;
            async_g2l_b128((unsigned)(size_t)&As[r * 64 + c8],
                           &A[(size_t)(mb + r) * Kd + k0 + c8]);
        }
        // B tile 64x64 bf16 -> transposed (N-major) scatter
#pragma unroll
        for (int i = 0; i < 4; ++i) {
            int g  = tid + i * 128;
            int r  = g >> 3;                  // k
            int c8 = (g & 7) * 8;             // n
            U4BF bb;
            bb.v = *(const uint4*)&Bw[(size_t)(k0 + r) * N + nb + c8];
#pragma unroll
            for (int j = 0; j < 8; ++j) BsT[(c8 + j) * BTLD + r] = bb.u[j];
        }
        wait_asynccnt0();
        __syncthreads();

#pragma unroll
        for (int ks = 0; ks < 64; ks += 32) {
            v16bf a0 = load_A_frag(&As[(32 * w) * 64 + ks], 64);
            v16bf a1 = load_A_frag(&As[(32 * w + 16) * 64 + ks], 64);
#pragma unroll
            for (int t = 0; t < 4; ++t) {
                v16bf b = load_BT_frag(&BsT[(16 * t) * BTLD + ks], BTLD);
                acc0[t] = wmma_bf16(a0, b, acc0[t]);
                acc1[t] = wmma_bf16(a1, b, acc1[t]);
            }
        }
    }

    const int col = lane & 15, hi = lane >> 4;
#pragma unroll
    for (int g16 = 0; g16 < 2; ++g16) {
#pragma unroll
        for (int t = 0; t < 4; ++t) {
            int n = nb + 16 * t + col;
#pragma unroll
            for (int r = 0; r < 8; ++r) {
                int mrow = mb + 32 * w + 16 * g16 + r + 8 * hi;
                float val = g16 ? acc1[t][r] : acc0[t][r];
                size_t oi = (size_t)mrow * N + n;
                if constexpr (EPI == 1) {           // bias + gelu (tanh approx)
                    val += bias[n];
                    float u = 0.7978845608f * (val + 0.044715f * val * val * val);
                    C[oi] = 0.5f * val * (1.0f + tanhf(u));
                } else if constexpr (EPI == 2) {    // residual add
                    C[oi] = resid[oi] + val;
                } else if constexpr (EPI == 3) {    // MoE weighted accumulate
                    float gg = gate[(size_t)mrow * Ec + gateIdx];
                    C[oi]   = C[oi] + gg * (val + bias[n]);
                } else {
                    C[oi] = val;
                }
            }
        }
    }
}

// ---------------------------------------------------------------------------
// Flash attention with WMMA. Grid (S/64, H, B), 128 threads (4 waves).
// ---------------------------------------------------------------------------
constexpr int VTLD = 40;   // padded hd-major V tile stride (u16)

__global__ __launch_bounds__(128) void attn_kernel(
    const float* __restrict__ q, const float* __restrict__ k,
    const float* __restrict__ v, float* __restrict__ ao)
{
    __shared__ unsigned short qs[64 * 64];       // [qrow][hd]
    __shared__ unsigned short ks[32 * 64];       // [key][hd]
    __shared__ unsigned short vsT[64 * VTLD];    // [hd][key]
    __shared__ unsigned short ps[4 * 16 * 32];   // per-wave P tile

    const int qb   = blockIdx.x * 64;
    const int h    = blockIdx.y;
    const int b    = blockIdx.z;
    const int tid  = threadIdx.x;
    const int lane = tid & 31;
    const int w    = tid >> 5;
    const int col  = lane & 15;
    const int hi   = lane >> 4;

#pragma unroll
    for (int i = 0; i < 8; ++i) {
        int g  = tid + i * 128;
        int r  = g >> 4;
        int c4 = (g & 15) * 4;
        const float4 qv = *(const float4*)&q[(((size_t)b * Sc + qb + r) * Hc + h) * HDc + c4];
        *(uint2*)&qs[r * 64 + c4] =
            make_uint2(f2bf2(qv.x * 0.125f, qv.y * 0.125f),
                       f2bf2(qv.z * 0.125f, qv.w * 0.125f));
    }
    __syncthreads();

    v16bf qA0 = load_A_frag(&qs[(16 * w) * 64], 64);
    v16bf qA1 = load_A_frag(&qs[(16 * w) * 64 + 32], 64);

    v8f oacc[4] = {};
    float m[8], l[8];
#pragma unroll
    for (int r = 0; r < 8; ++r) { m[r] = -1e30f; l[r] = 0.0f; }

    const int nsteps = (qb + 64) / 32;
    unsigned short* pw = &ps[w * 16 * 32];

    for (int step = 0; step < nsteps; ++step) {
        const int kb = step * 32;
        __syncthreads();
#pragma unroll
        for (int i = 0; i < 4; ++i) {
            int g  = tid + i * 128;
            int r  = g >> 4;                 // key
            int c4 = (g & 15) * 4;           // hd
            size_t gi = (((size_t)b * Sc + kb + r) * Hc + h) * HDc + c4;
            const float4 kv = *(const float4*)&k[gi];
            *(uint2*)&ks[r * 64 + c4] = make_uint2(f2bf2(kv.x, kv.y), f2bf2(kv.z, kv.w));
            const float4 vv = *(const float4*)&v[gi];
            vsT[(c4 + 0) * VTLD + r] = f2bf(vv.x);
            vsT[(c4 + 1) * VTLD + r] = f2bf(vv.y);
            vsT[(c4 + 2) * VTLD + r] = f2bf(vv.z);
            vsT[(c4 + 3) * VTLD + r] = f2bf(vv.w);
        }
        __syncthreads();

        v8f s0 = {}, s1 = {};
        s0 = wmma_bf16(qA0, load_KT_frag(ks, 0, 0),  s0);
        s0 = wmma_bf16(qA1, load_KT_frag(ks, 0, 32), s0);
        s1 = wmma_bf16(qA0, load_KT_frag(ks, 1, 0),  s1);
        s1 = wmma_bf16(qA1, load_KT_frag(ks, 1, 32), s1);

        float p0[8], p1[8], al[8];
#pragma unroll
        for (int r = 0; r < 8; ++r) {
            int qrow = qb + 16 * w + r + 8 * hi;
            int key0 = kb + col, key1 = kb + 16 + col;
            float x0 = (key0 <= qrow) ? s0[r] : -1e9f;
            float x1 = (key1 <= qrow) ? s1[r] : -1e9f;
            float mx = fmaxf(x0, x1);
#pragma unroll
            for (int msk = 8; msk >= 1; msk >>= 1)
                mx = fmaxf(mx, __shfl_xor(mx, msk, 32));
            float nm = fmaxf(m[r], mx);
            float a  = __expf(m[r] - nm);
            float e0 = __expf(x0 - nm);
            float e1 = __expf(x1 - nm);
            float rs = e0 + e1;
#pragma unroll
            for (int msk = 8; msk >= 1; msk >>= 1)
                rs += __shfl_xor(rs, msk, 32);
            l[r] = l[r] * a + rs;
            m[r] = nm;
            al[r] = a; p0[r] = e0; p1[r] = e1;
        }
#pragma unroll
        for (int t = 0; t < 4; ++t)
#pragma unroll
            for (int r = 0; r < 8; ++r) oacc[t][r] = oacc[t][r] * al[r];

#pragma unroll
        for (int r = 0; r < 8; ++r) {
            int row = r + 8 * hi;
            pw[row * 32 + col]      = f2bf(p0[r]);
            pw[row * 32 + 16 + col] = f2bf(p1[r]);
        }
        asm volatile("s_wait_dscnt 0x0" ::: "memory");
        v16bf pA = load_A_frag(pw, 32);
#pragma unroll
        for (int t = 0; t < 4; ++t) {
            v16bf vb = load_BT_frag(&vsT[(16 * t) * VTLD], VTLD);
            oacc[t] = wmma_bf16(pA, vb, oacc[t]);
        }
    }

#pragma unroll
    for (int t = 0; t < 4; ++t)
#pragma unroll
        for (int r = 0; r < 8; ++r) {
            int srow = qb + 16 * w + r + 8 * hi;
            float val = oacc[t][r] / l[r];
            ao[(size_t)(b * Sc + srow) * Dc + h * HDc + 16 * t + col] = val;
        }
}

// ---------------------------------------------------------------------------
// Small elementwise / reduction kernels
// ---------------------------------------------------------------------------
__global__ void embed_kernel(const int* __restrict__ ids,
                             const float* __restrict__ emb,
                             float* __restrict__ x, int total)
{
    int idx = blockIdx.x * blockDim.x + threadIdx.x;
    if (idx >= total) return;
    int t = idx / Dc, d = idx - t * Dc;
    x[idx] = emb[(size_t)ids[t] * Dc + d];
}

__global__ __launch_bounds__(128) void ln_kernel(
    const float* __restrict__ x, const float* __restrict__ g,
    const float* __restrict__ bb, float* __restrict__ out, int M)
{
    int t = blockIdx.x * 4 + (threadIdx.x >> 5);
    if (t >= M) return;
    int lane = threadIdx.x & 31;
    float s = 0.f, s2 = 0.f;
    for (int i = lane; i < Dc; i += 32) {
        float v = x[(size_t)t * Dc + i];
        s += v; s2 += v * v;
    }
#pragma unroll
    for (int msk = 16; msk >= 1; msk >>= 1) {
        s  += __shfl_xor(s, msk, 32);
        s2 += __shfl_xor(s2, msk, 32);
    }
    float mean = s / Dc;
    float var  = s2 / Dc - mean * mean;
    float inv  = rsqrtf(var + 1e-6f);
    for (int i = lane; i < Dc; i += 32)
        out[(size_t)t * Dc + i] = (x[(size_t)t * Dc + i] - mean) * inv * g[i] + bb[i];
}

__global__ void rope_kernel(float* __restrict__ x, int total)
{
    int idx = blockIdx.x * blockDim.x + threadIdx.x;   // pairs: B*S*H*32
    if (idx >= total) return;
    int i   = idx & 31;
    int bsh = idx >> 5;
    int s   = (bsh / Hc) % Sc;
    size_t base = (size_t)bsh * HDc;
    float inv_freq = __powf(10000.0f, -(float)i / 32.0f);
    float ang = (float)s * inv_freq;
    float c = __cosf(ang), sn = __sinf(ang);
    float x1 = x[base + i];
    float x2 = x[base + 32 + i];
    x[base + i]      = x1 * c - x2 * sn;
    x[base + 32 + i] = x2 * c + x1 * sn;
}

__global__ __launch_bounds__(128) void gate_kernel(
    const float* __restrict__ h2, const float* __restrict__ Wg,
    float* __restrict__ comb, int M)
{
    int t = blockIdx.x * 4 + (threadIdx.x >> 5);
    if (t >= M) return;
    int lane = threadIdx.x & 31;
    float logit[Ec];
#pragma unroll
    for (int e = 0; e < Ec; ++e) {
        float s = 0.f;
        for (int i = lane; i < Dc; i += 32)
            s += h2[(size_t)t * Dc + i] * Wg[(size_t)i * Ec + e];
#pragma unroll
        for (int msk = 16; msk >= 1; msk >>= 1) s += __shfl_xor(s, msk, 32);
        logit[e] = s;
    }
    float mx = logit[0];
#pragma unroll
    for (int e = 1; e < Ec; ++e) mx = fmaxf(mx, logit[e]);
    float p[Ec];
    float sum = 0.f;
#pragma unroll
    for (int e = 0; e < Ec; ++e) { p[e] = __expf(logit[e] - mx); sum += p[e]; }
#pragma unroll
    for (int e = 0; e < Ec; ++e) p[e] /= sum;
    int i0 = 0;
#pragma unroll
    for (int e = 1; e < Ec; ++e) if (p[e] > p[i0]) i0 = e;
    int i1 = (i0 == 0) ? 1 : 0;
#pragma unroll
    for (int e = 0; e < Ec; ++e) if (e != i0 && p[e] > p[i1]) i1 = e;
    float tot = p[i0] + p[i1];
    if (lane < Ec) {
        float wv = (lane == i0) ? p[i0] / tot : ((lane == i1) ? p[i1] / tot : 0.f);
        comb[(size_t)t * Ec + lane] = wv;
    }
}

// ---------------------------------------------------------------------------
// Launch sequence
// ---------------------------------------------------------------------------
static inline void cvt(const float* in, unsigned short* out, size_t n, hipStream_t s)
{
    int n8 = (int)(n / 8);
    cvt_kernel<<<(n8 + 255) / 256, 256, 0, s>>>(in, out, n8);
}

extern "C" void kernel_launch(void* const* d_in, const int* in_sizes, int n_in,
                              void* d_out, int out_size, void* d_ws, size_t ws_size,
                              hipStream_t stream)
{
    (void)in_sizes; (void)n_in; (void)out_size; (void)ws_size;

    const int*   ids   = (const int*)  d_in[0];
    const float* emb   = (const float*)d_in[1];
    const float* ln1_g = (const float*)d_in[2];
    const float* ln1_b = (const float*)d_in[3];
    const float* ln2_g = (const float*)d_in[4];
    const float* ln2_b = (const float*)d_in[5];
    const float* Wq    = (const float*)d_in[6];
    const float* Wkv   = (const float*)d_in[7];
    const float* Wk    = (const float*)d_in[8];
    const float* Wv    = (const float*)d_in[9];
    const float* Wo    = (const float*)d_in[10];
    const float* Wg    = (const float*)d_in[11];
    const float* W1    = (const float*)d_in[12];
    const float* b1    = (const float*)d_in[13];
    const float* W2    = (const float*)d_in[14];
    const float* b2    = (const float*)d_in[15];
    const float* lnf_g = (const float*)d_in[16];
    const float* lnf_b = (const float*)d_in[17];
    const float* Wout  = (const float*)d_in[18];
    float* out = (float*)d_out;

    const int M = Bc * Sc;   // 2048 tokens

    // f32 workspace
    float* ws   = (float*)d_ws;
    float* x    = ws;                  // M*Dc
    float* h    = x    + (size_t)M * Dc;
    float* lat  = h    + (size_t)M * Dc;
    float* qb   = lat  + (size_t)M * DLc;
    float* kb   = qb   + (size_t)M * Dc;
    float* vb   = kb   + (size_t)M * Dc;
    float* ao   = vb   + (size_t)M * Dc;
    float* h2   = ao   + (size_t)M * Dc;
    float* comb = h2   + (size_t)M * Dc;
    float* ff   = comb + (size_t)M * Ec;   // M*DFFc

    // bf16 workspace (after f32 region)
    unsigned short* bfb  = (unsigned short*)(ff + (size_t)M * DFFc);
    unsigned short* abfH = bfb;                          // M*Dc  (h / ao / h2)
    unsigned short* abfF = abfH + (size_t)M * Dc;        // M*DFFc (lat / ff)
    unsigned short* wbf  = abfF + (size_t)M * DFFc;      // up to Dc*Vc
    unsigned short* wbf2 = wbf  + (size_t)8 * 1024 * 1024;

    dim3 blk(128);

    embed_kernel<<<(M * Dc + 255) / 256, 256, 0, stream>>>(ids, emb, x, M * Dc);

    for (int l = 0; l < Lc; ++l) {
        ln_kernel<<<M / 4, blk, 0, stream>>>(x, ln1_g + l * Dc, ln1_b + l * Dc, h, M);
        cvt(h, abfH, (size_t)M * Dc, stream);
        cvt(Wq  + (size_t)l * Dc * Dc,  wbf,  (size_t)Dc * Dc,  stream);
        cvt(Wkv + (size_t)l * Dc * DLc, wbf2, (size_t)Dc * DLc, stream);

        gemm_kernel<0><<<dim3(Dc / 64, M / 128), blk, 0, stream>>>(
            abfH, wbf, qb, M, Dc, Dc, nullptr, nullptr, nullptr, 0);
        gemm_kernel<0><<<dim3(DLc / 64, M / 128), blk, 0, stream>>>(
            abfH, wbf2, lat, M, DLc, Dc, nullptr, nullptr, nullptr, 0);

        cvt(lat, abfF, (size_t)M * DLc, stream);
        cvt(Wk + (size_t)l * DLc * Dc, wbf,  (size_t)DLc * Dc, stream);
        cvt(Wv + (size_t)l * DLc * Dc, wbf2, (size_t)DLc * Dc, stream);

        gemm_kernel<0><<<dim3(Dc / 64, M / 128), blk, 0, stream>>>(
            abfF, wbf, kb, M, Dc, DLc, nullptr, nullptr, nullptr, 0);
        gemm_kernel<0><<<dim3(Dc / 64, M / 128), blk, 0, stream>>>(
            abfF, wbf2, vb, M, Dc, DLc, nullptr, nullptr, nullptr, 0);

        int rope_total = Bc * Sc * Hc * 32;
        rope_kernel<<<(rope_total + 255) / 256, 256, 0, stream>>>(qb, rope_total);
        rope_kernel<<<(rope_total + 255) / 256, 256, 0, stream>>>(kb, rope_total);

        attn_kernel<<<dim3(Sc / 64, Hc, Bc), blk, 0, stream>>>(qb, kb, vb, ao);

        // x = x + ao @ Wo
        cvt(ao, abfH, (size_t)M * Dc, stream);
        cvt(Wo + (size_t)l * Dc * Dc, wbf, (size_t)Dc * Dc, stream);
        gemm_kernel<2><<<dim3(Dc / 64, M / 128), blk, 0, stream>>>(
            abfH, wbf, x, M, Dc, Dc, nullptr, x, nullptr, 0);

        ln_kernel<<<M / 4, blk, 0, stream>>>(x, ln2_g + l * Dc, ln2_b + l * Dc, h2, M);
        cvt(h2, abfH, (size_t)M * Dc, stream);
        gate_kernel<<<M / 4, blk, 0, stream>>>(h2, Wg + (size_t)l * Dc * Ec, comb, M);

        for (int e = 0; e < Ec; ++e) {
            const float* W1e = W1 + ((size_t)(l * Ec + e)) * Dc * DFFc;
            const float* b1e = b1 + ((size_t)(l * Ec + e)) * DFFc;
            const float* W2e = W2 + ((size_t)(l * Ec + e)) * DFFc * Dc;
            const float* b2e = b2 + ((size_t)(l * Ec + e)) * Dc;
            cvt(W1e, wbf,  (size_t)Dc * DFFc, stream);
            cvt(W2e, wbf2, (size_t)DFFc * Dc, stream);
            // ff = gelu(h2 @ W1e + b1e)
            gemm_kernel<1><<<dim3(DFFc / 64, M / 128), blk, 0, stream>>>(
                abfH, wbf, ff, M, DFFc, Dc, b1e, nullptr, nullptr, 0);
            cvt(ff, abfF, (size_t)M * DFFc, stream);
            // x += comb[:,e] * (ff @ W2e + b2e)
            gemm_kernel<3><<<dim3(Dc / 64, M / 128), blk, 0, stream>>>(
                abfF, wbf2, x, M, Dc, DFFc, b2e, nullptr, comb, e);
        }
    }

    ln_kernel<<<M / 4, blk, 0, stream>>>(x, lnf_g, lnf_b, h, M);
    cvt(h, abfH, (size_t)M * Dc, stream);
    cvt(Wout, wbf, (size_t)Dc * Vc, stream);
    gemm_kernel<0><<<dim3(Vc / 64, M / 128), blk, 0, stream>>>(
        abfH, wbf, out, M, Vc, Dc, nullptr, nullptr, nullptr, 0);
}
